// Block_57793079935097
// MI455X (gfx1250) — compile-verified, matrix operand order checked by reading, not run on hardware
//
#include <hip/hip_runtime.h>

#define B_DIM   2
#define T_DIM   2048
#define C_DIM   1024
#define H_DIM   16
#define HD_DIM  64
#define DFF_DIM 4096
#define BT_DIM  (B_DIM * T_DIM)

typedef __attribute__((ext_vector_type(16))) __bf16 v16bf;
typedef __attribute__((ext_vector_type(8)))  float  v8f;
typedef __attribute__((ext_vector_type(4)))  unsigned int u32x4;

__device__ __forceinline__ v8f wmma_bf16(v16bf a, v16bf b, v8f c) {
  // D = A(16x32 bf16) * B(32x16 bf16) + C(16x16 f32)
  return __builtin_amdgcn_wmma_f32_16x16x32_bf16(
      /*neg_a=*/false, a, /*neg_b=*/false, b,
      /*c_mod=*/(short)0, c, /*reuse_a=*/false, /*reuse_b=*/false);
}

// Load one 16x32 bf16 fragment in the CDNA5 A-matrix lane layout from a
// row-major [16 x ld] tile.  Lanes 0-15: row=lane, K 0-7 (lo) / 16-23 (hi);
// lanes 16-31: row=lane-16, K 8-15 (lo) / 24-31 (hi).  Two b128 loads.
// Works for B as well when B is pre-transposed to [N][K].
__device__ __forceinline__ v16bf load_frag(const __bf16* base, int ld) {
  const int lane = threadIdx.x & 31;
  const int r    = lane & 15;
  const int kh   = (lane >> 4) << 3;        // 0 or 8
  const __bf16* p = base + (size_t)r * ld + kh;
  union { v16bf v; struct { u32x4 lo, hi; } q; } u;
  u.q.lo = *(const u32x4*)(p);
  u.q.hi = *(const u32x4*)(p + 16);
  return u.v;
}

__device__ __forceinline__ void wait_lds() {
  asm volatile("s_wait_dscnt 0" ::: "memory");
}

// ---------------------------------------------------------------- converts
__global__ __launch_bounds__(256) void f32_to_bf16_kernel(
    const float* __restrict__ in, __bf16* __restrict__ out, int n) {
  int i = (blockIdx.x * 256 + threadIdx.x) * 4;
  if (i + 3 < n) {
    float4 f = *(const float4*)(in + i);
    out[i + 0] = (__bf16)f.x;
    out[i + 1] = (__bf16)f.y;
    out[i + 2] = (__bf16)f.z;
    out[i + 3] = (__bf16)f.w;
  }
}

// in: f32 row-major [R][Cc] -> out: bf16 [Cc][R]   (weights -> [N][K] layout)
__global__ __launch_bounds__(256) void transpose_f32_to_bf16(
    const float* __restrict__ in, __bf16* __restrict__ out, int R, int Cc) {
  __shared__ float tile[32][33];
  const int c0 = blockIdx.x * 32, r0 = blockIdx.y * 32;
  for (int i = threadIdx.y; i < 32; i += 8)
    tile[i][threadIdx.x] = in[(size_t)(r0 + i) * Cc + c0 + threadIdx.x];
  __syncthreads();
  for (int i = threadIdx.y; i < 32; i += 8)
    out[(size_t)(c0 + i) * R + r0 + threadIdx.x] = (__bf16)tile[threadIdx.x][i];
}

// ---------------------------------------------------------------- GEMM
enum { EPI_QK = 0, EPI_VT = 1, EPI_F32_BIAS = 2, EPI_RELU_BF16 = 3 };

// C[M,N] = A[M,K](bf16) * Bt[N,K]^T(bf16), f32 accumulate.
// Block: 256 threads = 8 waves; block tile 256(M) x 64(N).
// Wave tile 32x64: 2 M-tiles x 4 N-tiles = 8 WMMAs per K=32 step, so each
// LDS B-fragment feeds two WMMAs.  B-tile (64x32) is double-buffered in LDS
// with a register prefetch of the next K-step, one barrier per step.
template <int EPI>
__global__ __launch_bounds__(256) void gemm_bf16_wmma(
    const __bf16* __restrict__ A, const __bf16* __restrict__ Bt,
    const float* __restrict__ bias, void* __restrict__ outp,
    int M, int N, int K) {
  __shared__ __bf16 Bs[2][64 * 32];

  const int lane = threadIdx.x & 31;
  const int wave = threadIdx.x >> 5;
  const int row0 = (blockIdx.y * 8 + wave) * 32;   // 32 rows per wave
  const int col0 = blockIdx.x * 64;

  v8f acc0[4] = {};
  v8f acc1[4] = {};

  const int ldn = threadIdx.x >> 2;         // 0..63 : local N row of Bs
  const int ldk = (threadIdx.x & 3) << 3;   // 0,8,16,24
  const __bf16* bsrc  = Bt + (size_t)(col0 + ldn) * K + ldk;
  const __bf16* asrc0 = A + (size_t)row0 * K;
  const __bf16* asrc1 = asrc0 + (size_t)16 * K;

  u32x4 breg = *(const u32x4*)(bsrc);       // prefetch K-step 0
  int p = 0;

  for (int k0 = 0; k0 < K; k0 += 32) {
    *(u32x4*)&Bs[p][ldn * 32 + ldk] = breg;
    __syncthreads();
    if (k0 + 32 < K) {
      breg = *(const u32x4*)(bsrc + k0 + 32);      // overlaps with WMMAs
      __builtin_prefetch(asrc0 + k0 + 32, 0, 1);
      __builtin_prefetch(asrc1 + k0 + 32, 0, 1);
    }
    v16bf a0 = load_frag(asrc0 + k0, K);
    v16bf a1 = load_frag(asrc1 + k0, K);
#pragma unroll
    for (int j = 0; j < 4; ++j) {
      v16bf bf = load_frag(&Bs[p][(j * 16) * 32], 32);
      acc0[j] = wmma_bf16(a0, bf, acc0[j]);
      acc1[j] = wmma_bf16(a1, bf, acc1[j]);
    }
    p ^= 1;   // next store goes to the other buffer; no 2nd barrier needed
  }

  const int half = lane >> 4, ln = lane & 15;
#pragma unroll
  for (int m2 = 0; m2 < 2; ++m2) {
#pragma unroll
    for (int j = 0; j < 4; ++j) {
#pragma unroll
      for (int r = 0; r < 8; ++r) {
        const int row = row0 + m2 * 16 + half * 8 + r;  // D layout
        const int col = col0 + j * 16 + ln;
        float v = m2 ? acc1[j][r] : acc0[j][r];
        if constexpr (EPI == EPI_QK) {
          // [BT,C] -> bf16 [B,H,T,hd]
          const int bi = row >> 11, ti = row & (T_DIM - 1);
          const int hi = col >> 6, di = col & (HD_DIM - 1);
          ((__bf16*)outp)[(((size_t)bi * H_DIM + hi) * T_DIM + ti) * HD_DIM + di] =
              (__bf16)v;
        } else if constexpr (EPI == EPI_VT) {
          // [BT,C] -> bf16 [B,H,hd,T]  (V transposed: K-contiguous for P*V)
          const int bi = row >> 11, ti = row & (T_DIM - 1);
          const int hi = col >> 6, di = col & (HD_DIM - 1);
          ((__bf16*)outp)[(((size_t)bi * H_DIM + hi) * HD_DIM + di) * T_DIM + ti] =
              (__bf16)v;
        } else if constexpr (EPI == EPI_F32_BIAS) {
          ((float*)outp)[(size_t)row * N + col] = v + bias[col];
        } else {  // EPI_RELU_BF16
          float o = v + bias[col];
          o = o > 0.f ? o : 0.f;
          ((__bf16*)outp)[(size_t)row * N + col] = (__bf16)o;
        }
      }
    }
  }
}

// ---------------------------------------------------------------- attention
// Flash-style causal attention. One wave per 16-row Q tile; K blocks of 32.
#define ATT_WAVES 4
__global__ __launch_bounds__(128) void attn_wmma_kernel(
    const __bf16* __restrict__ q,   // [B,H,T,hd]
    const __bf16* __restrict__ k,   // [B,H,T,hd]
    const __bf16* __restrict__ vt,  // [B,H,hd,T]
    __bf16* __restrict__ attnb) {   // bf16 [B,T,C]
  __shared__ __bf16 lds[ATT_WAVES][16 * 32];

  const int wave = threadIdx.x >> 5;
  const int lane = threadIdx.x & 31;
  const int half = lane >> 4, ln = lane & 15;

  const int qt0 = (blockIdx.x * ATT_WAVES + wave) * 16;
  const int bh  = blockIdx.y;
  const int bi  = bh >> 4;          // /H
  const int hi  = bh & (H_DIM - 1);

  const __bf16* Qp = q + ((size_t)bh * T_DIM + qt0) * HD_DIM;
  const __bf16* Kp = k + (size_t)bh * T_DIM * HD_DIM;
  const __bf16* Vp = vt + (size_t)bh * HD_DIM * T_DIM;

  const v16bf qa0 = load_frag(Qp, HD_DIM);        // K = 0..31 of hd
  const v16bf qa1 = load_frag(Qp + 32, HD_DIM);   // K = 32..63 of hd

  v8f o[4] = {};
  float rm[8], rl[8];
#pragma unroll
  for (int r = 0; r < 8; ++r) { rm[r] = -1e30f; rl[r] = 0.f; }

  __bf16* L = &lds[wave][0];

  for (int kv0 = 0; kv0 < qt0 + 16; kv0 += 32) {
    float pm[2][8];
#pragma unroll
    for (int j = 0; j < 2; ++j) {
      const __bf16* Kb = Kp + (size_t)(kv0 + j * 16) * HD_DIM;
      v16bf kb0 = load_frag(Kb, HD_DIM);
      v16bf kb1 = load_frag(Kb + 32, HD_DIM);
      v8f s = {};
      s = wmma_bf16(qa0, kb0, s);
      s = wmma_bf16(qa1, kb1, s);
#pragma unroll
      for (int r = 0; r < 8; ++r) {
        float val = s[r] * 0.125f;              // hd^-0.5
        const int qg = qt0 + half * 8 + r;
        const int kg = kv0 + j * 16 + ln;
        pm[j][r] = (kg > qg) ? -1e30f : val;    // causal mask
      }
    }
    // online softmax, rows live in VGPR index r within each 16-lane half
#pragma unroll
    for (int r = 0; r < 8; ++r) {
      float mx = fmaxf(pm[0][r], pm[1][r]);
      mx = fmaxf(mx, __shfl_xor(mx, 1, 32));
      mx = fmaxf(mx, __shfl_xor(mx, 2, 32));
      mx = fmaxf(mx, __shfl_xor(mx, 4, 32));
      mx = fmaxf(mx, __shfl_xor(mx, 8, 32));
      const float mnew  = fmaxf(rm[r], mx);
      const float scale = __expf(rm[r] - mnew);
      rm[r] = mnew;
      const float p0 = __expf(pm[0][r] - mnew);
      const float p1 = __expf(pm[1][r] - mnew);
      float rs = p0 + p1;
      rs += __shfl_xor(rs, 1, 32);
      rs += __shfl_xor(rs, 2, 32);
      rs += __shfl_xor(rs, 4, 32);
      rs += __shfl_xor(rs, 8, 32);
      rl[r] = rl[r] * scale + rs;
      pm[0][r] = p0; pm[1][r] = p1;
#pragma unroll
      for (int jj = 0; jj < 4; ++jj) o[jj][r] *= scale;
    }
    // P (16x32, D-layout) -> LDS -> A-fragment layout
#pragma unroll
    for (int j = 0; j < 2; ++j)
#pragma unroll
      for (int r = 0; r < 8; ++r)
        L[(half * 8 + r) * 32 + j * 16 + ln] = (__bf16)pm[j][r];
    wait_lds();
    const v16bf pf = load_frag(L, 32);
#pragma unroll
    for (int jj = 0; jj < 4; ++jj) {
      v16bf vb = load_frag(Vp + (size_t)(jj * 16) * T_DIM + kv0, T_DIM);
      o[jj] = wmma_bf16(pf, vb, o[jj]);
    }
  }
  // normalize + scatter back to [B,T,C] bf16
#pragma unroll
  for (int r = 0; r < 8; ++r) {
    const float inv = 1.f / rl[r];
    const int qg = qt0 + half * 8 + r;
#pragma unroll
    for (int jj = 0; jj < 4; ++jj) {
      size_t dst = ((size_t)bi * T_DIM + qg) * C_DIM + hi * HD_DIM + jj * 16 + ln;
      attnb[dst] = (__bf16)(o[jj][r] * inv);
    }
  }
}

// ---------------------------------------------------------------- LN
// out = LN(x + y) * g + b ; one block per row of C_DIM=1024, 256 threads.
__global__ __launch_bounds__(256) void ln_residual_kernel(
    const float* __restrict__ x, const float* __restrict__ y,
    const float* __restrict__ g, const float* __restrict__ b,
    float* __restrict__ out, __bf16* __restrict__ outb) {
  const int row = blockIdx.x;
  const float* xr = x + (size_t)row * C_DIM;
  const float* yr = y + (size_t)row * C_DIM;
  float v[4], s = 0.f, s2 = 0.f;
#pragma unroll
  for (int i = 0; i < 4; ++i) {
    const int idx = threadIdx.x + i * 256;
    const float t = xr[idx] + yr[idx];
    v[i] = t; s += t; s2 += t * t;
  }
#pragma unroll
  for (int m = 1; m < 32; m <<= 1) {
    s  += __shfl_xor(s, m, 32);
    s2 += __shfl_xor(s2, m, 32);
  }
  __shared__ float w1s[8], w2s[8];
  if ((threadIdx.x & 31) == 0) {
    w1s[threadIdx.x >> 5] = s;
    w2s[threadIdx.x >> 5] = s2;
  }
  __syncthreads();
  s = 0.f; s2 = 0.f;
#pragma unroll
  for (int j = 0; j < 8; ++j) { s += w1s[j]; s2 += w2s[j]; }
  const float mean = s * (1.f / C_DIM);
  const float var  = s2 * (1.f / C_DIM) - mean * mean;
  const float rstd = rsqrtf(var + 1e-5f);
#pragma unroll
  for (int i = 0; i < 4; ++i) {
    const int idx = threadIdx.x + i * 256;
    const float o = (v[i] - mean) * rstd * g[idx] + b[idx];
    out[(size_t)row * C_DIM + idx] = o;
    if (outb) outb[(size_t)row * C_DIM + idx] = (__bf16)o;
  }
}

// ---------------------------------------------------------------- launch
extern "C" void kernel_launch(void* const* d_in, const int* in_sizes, int n_in,
                              void* d_out, int out_size, void* d_ws, size_t ws_size,
                              hipStream_t stream) {
  const float* x     = (const float*)d_in[0];
  const float* wq    = (const float*)d_in[1];
  const float* wk    = (const float*)d_in[2];
  const float* wv    = (const float*)d_in[3];
  const float* wp    = (const float*)d_in[4];
  const float* bproj = (const float*)d_in[5];
  const float* w1    = (const float*)d_in[6];
  const float* b1    = (const float*)d_in[7];
  const float* w2    = (const float*)d_in[8];
  const float* b2    = (const float*)d_in[9];
  const float* g1    = (const float*)d_in[10];
  const float* be1   = (const float*)d_in[11];
  const float* g2    = (const float*)d_in[12];
  const float* be2   = (const float*)d_in[13];

  char* ws = (char*)d_ws;
  size_t o = 0;
  auto take = [&](size_t nbytes) -> void* {
    void* p = ws + o;
    o += (nbytes + 255) & ~(size_t)255;
    return p;
  };
  const size_t BTC = (size_t)BT_DIM * C_DIM;
  __bf16* xb    = (__bf16*)take(BTC * 2);
  __bf16* wqt   = (__bf16*)take((size_t)C_DIM * C_DIM * 2);
  __bf16* wkt   = (__bf16*)take((size_t)C_DIM * C_DIM * 2);
  __bf16* wvt   = (__bf16*)take((size_t)C_DIM * C_DIM * 2);
  __bf16* wpt   = (__bf16*)take((size_t)C_DIM * C_DIM * 2);
  __bf16* w1t   = (__bf16*)take((size_t)C_DIM * DFF_DIM * 2);
  __bf16* w2t   = (__bf16*)take((size_t)DFF_DIM * C_DIM * 2);
  __bf16* qb    = (__bf16*)take(BTC * 2);
  __bf16* kb    = (__bf16*)take(BTC * 2);
  __bf16* vtb   = (__bf16*)take(BTC * 2);
  __bf16* attnb = (__bf16*)take(BTC * 2);
  float*  y1    = (float*)take(BTC * 4);
  float*  x1    = (float*)take(BTC * 4);
  __bf16* x1b   = (__bf16*)take(BTC * 2);
  __bf16* h1    = (__bf16*)take((size_t)BT_DIM * DFF_DIM * 2);
  float*  y2    = (float*)take(BTC * 4);

  const dim3 tb(32, 8);
  f32_to_bf16_kernel<<<(BTC / 4 + 255) / 256, 256, 0, stream>>>(x, xb, (int)BTC);
  transpose_f32_to_bf16<<<dim3(C_DIM / 32, C_DIM / 32), tb, 0, stream>>>(wq, wqt, C_DIM, C_DIM);
  transpose_f32_to_bf16<<<dim3(C_DIM / 32, C_DIM / 32), tb, 0, stream>>>(wk, wkt, C_DIM, C_DIM);
  transpose_f32_to_bf16<<<dim3(C_DIM / 32, C_DIM / 32), tb, 0, stream>>>(wv, wvt, C_DIM, C_DIM);
  transpose_f32_to_bf16<<<dim3(C_DIM / 32, C_DIM / 32), tb, 0, stream>>>(wp, wpt, C_DIM, C_DIM);
  transpose_f32_to_bf16<<<dim3(DFF_DIM / 32, C_DIM / 32), tb, 0, stream>>>(w1, w1t, C_DIM, DFF_DIM);
  transpose_f32_to_bf16<<<dim3(C_DIM / 32, DFF_DIM / 32), tb, 0, stream>>>(w2, w2t, DFF_DIM, C_DIM);

  // block tile 256(M) x 64(N)
  const dim3 gC(C_DIM / 64, BT_DIM / 256);
  gemm_bf16_wmma<EPI_QK><<<gC, 256, 0, stream>>>(xb, wqt, nullptr, qb, BT_DIM, C_DIM, C_DIM);
  gemm_bf16_wmma<EPI_QK><<<gC, 256, 0, stream>>>(xb, wkt, nullptr, kb, BT_DIM, C_DIM, C_DIM);
  gemm_bf16_wmma<EPI_VT><<<gC, 256, 0, stream>>>(xb, wvt, nullptr, vtb, BT_DIM, C_DIM, C_DIM);

  attn_wmma_kernel<<<dim3(T_DIM / (16 * ATT_WAVES), B_DIM * H_DIM), 128, 0, stream>>>(
      qb, kb, vtb, attnb);

  gemm_bf16_wmma<EPI_F32_BIAS><<<gC, 256, 0, stream>>>(attnb, wpt, bproj, y1, BT_DIM, C_DIM, C_DIM);
  ln_residual_kernel<<<BT_DIM, 256, 0, stream>>>(x, y1, g1, be1, x1, x1b);

  gemm_bf16_wmma<EPI_RELU_BF16><<<dim3(DFF_DIM / 64, BT_DIM / 256), 256, 0, stream>>>(
      x1b, w1t, b1, h1, BT_DIM, DFF_DIM, C_DIM);
  gemm_bf16_wmma<EPI_F32_BIAS><<<gC, 256, 0, stream>>>(h1, w2t, b2, y2, BT_DIM, C_DIM, DFF_DIM);
  ln_residual_kernel<<<BT_DIM, 256, 0, stream>>>(x1, y2, g2, be2, (float*)d_out, nullptr);
}
